// GraphAttentionLayer_83056077570372
// MI455X (gfx1250) — matrix operand order repaired and checked by reading, not compiled, not run on hardware
//
#include <hip/hip_runtime.h>
#include <hip/hip_bf16.h>

// GAT fused kernels for MI455X (gfx1250), wave32 + WMMA + TDM.
// Shapes fixed per reference: B=8, N=1024, IN_F=128, OUT_F=256, H=4, D=64.

typedef __attribute__((ext_vector_type(16))) _Float16     v16h;
typedef __attribute__((ext_vector_type(8)))  float        v8f;
typedef __attribute__((ext_vector_type(4)))  unsigned int v4u;
typedef __attribute__((ext_vector_type(8)))  int          v8i;
typedef __attribute__((ext_vector_type(4)))  int          v4i;

#define GAT_B 8
#define GAT_N 1024
#define GAT_INF 128
#define GAT_OUTF 256
#define GAT_H 4
#define GAT_D 64
#define NEG_SLOPE 0.2f
#define LOG2E 1.4426950408889634f

#if defined(__has_builtin)
#if __has_builtin(__builtin_amdgcn_tensor_load_to_lds) && \
    __has_builtin(__builtin_amdgcn_s_wait_tensorcnt)
#define GAT_HAS_TDM 1
#endif
#endif
#ifndef GAT_HAS_TDM
#define GAT_HAS_TDM 0
#endif

// --- raw cross-lane helpers (avoid __shfl's bounds-check cndmask chains) ----
template <int PAT>
__device__ __forceinline__ float swz_xor(float v)   // ds_swizzle xor pattern
{
    return __int_as_float(__builtin_amdgcn_ds_swizzle(__float_as_int(v), PAT));
}
__device__ __forceinline__ float lane_bcast(float v, int srcLane)
{
    return __int_as_float(
        __builtin_amdgcn_ds_bpermute(srcLane << 2, __float_as_int(v)));
}
#define SWZ_XOR1  0x041f
#define SWZ_XOR2  0x081f
#define SWZ_XOR4  0x101f
#define SWZ_XOR8  0x201f
#define SWZ_XOR16 0x401f

#if GAT_HAS_TDM
// Issue one TDM 1-D tile copy: nelems f32 from global -> LDS (ISA 08 §8 D#).
__device__ __forceinline__ void tdm_load_row_f32(const float* gsrc, void* lds_dst,
                                                 int nelems)
{
    unsigned long long ga = (unsigned long long)(const void*)gsrc;
    v4u g0;
    g0.x = 1u;                                            // count=1 (valid user D#)
    g0.y = (unsigned int)(size_t)lds_dst;                 // lds_addr (bytes)
    g0.z = (unsigned int)(ga & 0xffffffffu);              // global_addr[31:0]
    g0.w = (unsigned int)((ga >> 32) & 0x1ffffffu)        // global_addr[56:32]
           | (2u << 30);                                  // type=2 ("image")
    v8i g1 = {};
    g1[0] = (2 << 16);                                    // data_size = 4 bytes
    g1[1] = (nelems & 0xffff) << 16;                      // tensor_dim0[15:0]
    g1[2] = (nelems >> 16) & 0xffff;                      // tensor_dim0[31:16]
    g1[3] = (nelems & 0xffff) << 16;                      // tile_dim0
    v4i z4 = {};
#if __clang_major__ >= 23
    v8i z8 = {};
    __builtin_amdgcn_tensor_load_to_lds(g0, g1, z4, z4, z8, 0);
#else
    __builtin_amdgcn_tensor_load_to_lds(g0, g1, z4, z4, 0);
#endif
}
#endif

// ---------------------------------------------------------------------------
// Pre-pass 1: pack adjacency int32 [B*N][N] -> bitmask [B*N][N/32] via wave32
// __ballot. Pure streaming pass carrying the only real HBM traffic (33.5 MB).
// ---------------------------------------------------------------------------
__global__ __launch_bounds__(256)
void gat_pack_adj(const int* __restrict__ adj, unsigned int* __restrict__ packed)
{
    const int ln   = threadIdx.x & 31;
    const int wv   = threadIdx.x >> 5;                 // 8 waves/block
    const int row  = blockIdx.x * 8 + wv;              // [0, B*N)
    const int* arow = adj + (size_t)row * GAT_N;
    unsigned int* prow = packed + (size_t)row * (GAT_N / 32);
#pragma unroll 4
    for (int jt = 0; jt < GAT_N / 32; ++jt) {
        const int v = arow[jt * 32 + ln];
        const unsigned long long bal = __ballot(v != 0);   // wave32: bits 0..31
        if (ln == 0) prow[jt] = (unsigned int)bal;
    }
}

// ---------------------------------------------------------------------------
// Pre-pass 2: W f32 [128][256] -> WT f16 [256][128] so projection B-fragments
// are single contiguous 32-byte v16h loads.
// ---------------------------------------------------------------------------
__global__ __launch_bounds__(256)
void gat_prep_w(const float* __restrict__ W, _Float16* __restrict__ WT)
{
    const int idx = blockIdx.x * 256 + threadIdx.x;    // 32768 elements
    const int c = idx >> 7, k = idx & 127;
    WT[(size_t)c * GAT_INF + k] = (_Float16)W[(size_t)k * GAT_OUTF + c];
}

// ---------------------------------------------------------------------------
// Stage 1: h = X @ W (f16 WMMA, f32 accum), producing
//   hT  : f16 [B][H][64][1024] (transposed for stage-2 B-frags)
//   s_i/s_j : f32 [B][H][N] head-wise dots with a[:64]/a[64:], PRE-SCALED by
//             log2(e) so stage 2 runs softmax in exp2 domain (LeakyReLU
//             commutes with positive scaling).
// One wave per 16-row slab of [B*N, 128].
// ---------------------------------------------------------------------------
__global__ __launch_bounds__(32)
void gat_proj(const float* __restrict__ X, const _Float16* __restrict__ WT,
              const float* __restrict__ a,
              _Float16* __restrict__ hT,
              float* __restrict__ s_i, float* __restrict__ s_j)
{
    const int ln = threadIdx.x;
    const int r0 = blockIdx.x * 16;
    const int m  = ln & 15;
    const int hi = ln >> 4;

    // A-fragments 16x32 f16: lanes 0-15 K=0..15, lanes 16-31 K=16..31 (contiguous)
    v16h afrag[4];
    const float* xrow = X + (size_t)(r0 + m) * GAT_INF;
#pragma unroll
    for (int ks = 0; ks < 4; ++ks) {
        const int k0 = ks * 32 + hi * 16;
#pragma unroll
        for (int i = 0; i < 8; ++i) {
            afrag[ks][i]     = (_Float16)xrow[k0 + i];
            afrag[ks][8 + i] = (_Float16)xrow[k0 + 8 + i];
        }
    }

#pragma unroll
    for (int head = 0; head < GAT_H; ++head) {
        float si[8], sj[8];
#pragma unroll
        for (int g = 0; g < 8; ++g) { si[g] = 0.f; sj[g] = 0.f; }

#pragma unroll
        for (int dt = 0; dt < 4; ++dt) {
            const int c0 = head * GAT_D + dt * 16;
            const _Float16* wcol = WT + (size_t)(c0 + m) * GAT_INF;
            v8f c = {};
#pragma unroll
            for (int ks = 0; ks < 4; ++ks) {
                // B-frag 32x16: lanes 0-15 K=0..15, lanes 16-31 K=16..31 (contiguous)
                v16h bfrag = *(const v16h*)(wcol + ks * 32 + hi * 16);
                c = __builtin_amdgcn_wmma_f32_16x16x32_f16(
                        false, afrag[ks], false, bfrag, (short)0, c, false, false);
            }

            const int d  = dt * 16 + m;
            const float a1 = a[d];
            const float a2 = a[GAT_D + d];
            // per-lane partials only; cross-lane reduction hoisted out of dt loop
#pragma unroll
            for (int g = 0; g < 8; ++g) {
                const float v  = c[g];
                const int   gr = r0 + (hi ? g + 8 : g);
                const int   b  = gr >> 10;
                const int   n  = gr & 1023;
                hT[(((size_t)(b * GAT_H + head) * GAT_D) + d) * GAT_N + n] = (_Float16)v;
                si[g] += v * a1;
                sj[g] += v * a2;
            }
        }

        // one xor-tree per head per g (stays inside the 16-lane half)
#pragma unroll
        for (int g = 0; g < 8; ++g) {
            si[g] += swz_xor<SWZ_XOR1>(si[g]);  sj[g] += swz_xor<SWZ_XOR1>(sj[g]);
            si[g] += swz_xor<SWZ_XOR2>(si[g]);  sj[g] += swz_xor<SWZ_XOR2>(sj[g]);
            si[g] += swz_xor<SWZ_XOR4>(si[g]);  sj[g] += swz_xor<SWZ_XOR4>(sj[g]);
            si[g] += swz_xor<SWZ_XOR8>(si[g]);  sj[g] += swz_xor<SWZ_XOR8>(sj[g]);
        }
        if (m == 0) {
#pragma unroll
            for (int g = 0; g < 8; ++g) {
                const int gr = r0 + (hi ? g + 8 : g);
                const int b = gr >> 10, n = gr & 1023;
                s_i[((size_t)b * GAT_H + head) * GAT_N + n] = si[g] * LOG2E;
                s_j[((size_t)b * GAT_H + head) * GAT_N + n] = sj[g] * LOG2E;
            }
        }
    }
}

// ---------------------------------------------------------------------------
// Stage 2: fused mask + leakyrelu + online softmax (exp2-domain) + WMMA
// aggregation. grid = B*(N/16); 4 waves/block (one head each).
// acc[0..3]: 16x64 numerator; acc[4]: P @ ones -> softmax denominator in the
// same C layout (no shuffles for l). s_j staged in LDS via TDM; adjacency
// from packed bits.
// ---------------------------------------------------------------------------
__global__ __launch_bounds__(128)
void gat_attn(const unsigned int* __restrict__ packed,
              const _Float16* __restrict__ hT,
              const float* __restrict__ s_i, const float* __restrict__ s_j,
              float* __restrict__ out)
{
    const int ln   = threadIdx.x & 31;
    const int head = threadIdx.x >> 5;
    const int b    = blockIdx.x >> 6;
    const int i0   = (blockIdx.x & 63) * 16;
    const int m    = ln & 15;
    const int hi   = ln >> 4;

    __shared__ float sj_lds[GAT_H * GAT_N];            // 16 KB
    float* sjs = sj_lds + head * GAT_N;
    const float* sjrow = s_j + ((size_t)b * GAT_H + head) * GAT_N;

#if GAT_HAS_TDM
    // one TDM descriptor per wave: 4 KB s_j row -> LDS, tracked by TENSORcnt
    tdm_load_row_f32(sjrow, sjs, GAT_N);
    __builtin_amdgcn_s_wait_tensorcnt(0);
#else
    for (int t = ln; t < GAT_N / 4; t += 32)
        ((float4*)sjs)[t] = ((const float4*)sjrow)[t];
#endif

    const float siv = s_i[((size_t)b * GAT_H + head) * GAT_N + i0 + m];
    const unsigned int* prow = packed + (size_t)(b * GAT_N + i0 + m) * (GAT_N / 32);
    const _Float16* hTh = hT + ((size_t)(b * GAT_H + head) * GAT_D) * GAT_N;

    v16h ones;
#pragma unroll
    for (int i = 0; i < 16; ++i) ones[i] = (_Float16)1.0f;

    float m_run = -1e30f;
    v8f acc[5] = {{}, {}, {}, {}, {}};

    for (int jt = 0; jt < GAT_N / 32; ++jt) {
        const int j0   = jt * 32;
        const int base = j0 + hi * 16;                 // this lane's K-group start

        if (jt + 1 < GAT_N / 32)
            __builtin_prefetch(prow + jt + 1, 0, 0);   // global_prefetch_b8

        const unsigned int pk = prow[jt];              // 32 adjacency bits

        // s_j from LDS, matching the f16 A-frag K layout:
        // elements 0..15 -> j=base+i
        alignas(16) float sv[16];
        *(float4*)(sv + 0)  = *(const float4*)(sjs + base);
        *(float4*)(sv + 4)  = *(const float4*)(sjs + base + 4);
        *(float4*)(sv + 8)  = *(const float4*)(sjs + base + 8);
        *(float4*)(sv + 12) = *(const float4*)(sjs + base + 12);

        // masked leakyrelu scores (already in exp2 domain)
        float tm[16];
        float tmax = -1e30f;
#pragma unroll
        for (int i = 0; i < 16; ++i) {
            const int off = hi * 16 + i;                    // bit within 32-tile
            const float x = siv + sv[i];
            const float e = fmaxf(x, NEG_SLOPE * x);        // LeakyReLU (scale>0)
            tm[i] = ((pk >> off) & 1u) ? e : -1e30f;
            tmax  = fmaxf(tmax, tm[i]);
        }
        tmax = fmaxf(tmax, swz_xor<SWZ_XOR16>(tmax));       // both row halves
        const float m_new = fmaxf(m_run, tmax);
        const float scale = exp2f(m_run - m_new);
        m_run = m_new;

        // P tile: exp2 underflows to exactly 0 for masked entries
        v16h pf;
#pragma unroll
        for (int i = 0; i < 16; ++i)
            pf[i] = (_Float16)exp2f(tm[i] - m_new);

        // rescale only when some row's max actually moved (wave-uniform branch)
        if (__any(scale < 1.f)) {
            float rs[8];
#pragma unroll
            for (int g = 0; g < 8; ++g)
                rs[g] = lane_bcast(scale, hi ? g + 8 : g);
#pragma unroll
            for (int t = 0; t < 5; ++t)
#pragma unroll
                for (int g = 0; g < 8; ++g)
                    acc[t][g] *= rs[g];
        }

#pragma unroll
        for (int dt = 0; dt < 4; ++dt) {
            const _Float16* bp = hTh + (size_t)(dt * 16 + m) * GAT_N + j0 + hi * 16;
            v16h bf = *(const v16h*)bp;                // 32B aligned, 2x b128
            acc[dt] = __builtin_amdgcn_wmma_f32_16x16x32_f16(
                          false, pf, false, bf, (short)0, acc[dt], false, false);
        }
        // denominator: P @ ones -> every column holds the row sum
        acc[4] = __builtin_amdgcn_wmma_f32_16x16x32_f16(
                     false, pf, false, ones, (short)0, acc[4], false, false);
    }

    // epilogue: l is already in C layout -> purely lane-local normalize
    float li[8];
#pragma unroll
    for (int g = 0; g < 8; ++g) {
        const float l = acc[4][g];
        li[g] = (l > 0.f) ? 1.f / l : 0.f;
    }
#pragma unroll
    for (int dt = 0; dt < 4; ++dt)
#pragma unroll
        for (int g = 0; g < 8; ++g) {
            const int row = i0 + (hi ? g + 8 : g);
            out[(size_t)(b * GAT_N + row) * GAT_OUTF + head * GAT_D + dt * 16 + m] =
                acc[dt][g] * li[g];
        }
}

// ---------------------------------------------------------------------------
extern "C" void kernel_launch(void* const* d_in, const int* in_sizes, int n_in,
                              void* d_out, int out_size, void* d_ws, size_t ws_size,
                              hipStream_t stream)
{
    const float* X   = (const float*)d_in[0];   // [8,1024,128]
    const int*   adj = (const int*)d_in[1];     // [8,1024,1024]
    const float* W   = (const float*)d_in[2];   // [128,256]
    const float* a   = (const float*)d_in[3];   // [128]
    float* out = (float*)d_out;                 // [8,1024,256]

    // workspace layout
    constexpr size_t HT_BYTES = (size_t)GAT_B * GAT_H * GAT_D * GAT_N * 2; // 4 MiB
    constexpr size_t S_BYTES  = (size_t)GAT_B * GAT_H * GAT_N * 4;         // 128 KiB
    constexpr size_t PK_BYTES = (size_t)GAT_B * GAT_N * (GAT_N / 32) * 4;  // 1 MiB

    char* ws = (char*)d_ws;
    _Float16*     hT     = (_Float16*)ws;
    float*        s_i    = (float*)(ws + HT_BYTES);
    float*        s_j    = (float*)(ws + HT_BYTES + S_BYTES);
    unsigned int* packed = (unsigned int*)(ws + HT_BYTES + 2 * S_BYTES);
    _Float16*     WT     = (_Float16*)(ws + HT_BYTES + 2 * S_BYTES + PK_BYTES);

    gat_pack_adj<<<(GAT_B * GAT_N) / 8, 256, 0, stream>>>(adj, packed);
    gat_prep_w<<<(GAT_OUTF * GAT_INF) / 256, 256, 0, stream>>>(W, WT);
    gat_proj<<<(GAT_B * GAT_N) / 16, 32, 0, stream>>>(X, WT, a, hT, s_i, s_j);
    gat_attn<<<GAT_B * (GAT_N / 16), 128, 0, stream>>>(packed, hT, s_i, s_j, out);
}
